// _LoRA_qkv_timm_eval_8555574853761
// MI455X (gfx1250) — matrix operand order repaired and checked by reading, not compiled, not run on hardware
//
#include <hip/hip_runtime.h>

// ---------------- problem constants (fixed by the reference) ----------------
constexpr int kD   = 768;        // D
constexpr int kN   = 2304;       // 3*D
constexpr int kM   = 65536;      // B*S
constexpr int kR   = 4;          // LoRA rank
constexpr int LDA  = 40;         // LDS row stride in halves (80B, conflict-free, 16B aligned)

typedef __attribute__((ext_vector_type(16))) __bf16 v16bf;
typedef __attribute__((ext_vector_type(8)))  float  v8f;
typedef __attribute__((ext_vector_type(4)))  int    v4i;

union FragAB { uint4 u[2]; v16bf v; };

__device__ __forceinline__ unsigned short f2bf(float f) {
    unsigned u = __float_as_uint(f);
    u += 0x7FFFu + ((u >> 16) & 1u);       // round-to-nearest-even
    return (unsigned short)(u >> 16);
}

// -------- CDNA5 async global->LDS copy (ASYNCcnt path), with fallback -------
#if __has_builtin(__builtin_amdgcn_global_load_async_to_lds_b128)
#define HAVE_ASYNC_LDS 1
// pointer-to-AS1/AS3 int4 (attribute binds to the pointee type)
typedef __attribute__((address_space(1))) v4i* as1_v4i;
typedef __attribute__((address_space(3))) v4i* as3_v4i;

__device__ __forceinline__ void async_b128(void* lds, const void* g) {
    // AS1 shares the flat 64-bit numeric address space -> integer round-trip is exact.
    // Generic->AS3 must be a pointer cast (addrspacecast extracts the LDS offset).
    __builtin_amdgcn_global_load_async_to_lds_b128(
        (as1_v4i)(unsigned long long)g,
        (as3_v4i)lds,
        /*offset=*/0, /*cpol=*/0);
}
__device__ __forceinline__ void wait_async0() {
#if __has_builtin(__builtin_amdgcn_s_wait_asynccnt)
    __builtin_amdgcn_s_wait_asynccnt(0);
#else
    asm volatile("s_wait_asynccnt 0x0" ::: "memory");
#endif
}
#else
#define HAVE_ASYNC_LDS 0
#endif

// ---------------- kernel 1: f32 -> bf16 pack (4 elements / thread) ----------
__global__ void k_convert_bf16(const float* __restrict__ src,
                               unsigned short* __restrict__ dst) {
    size_t i = ((size_t)blockIdx.x * blockDim.x + threadIdx.x) * 4;
    float4 f = *(const float4*)(src + i);
    ushort4 o;
    o.x = f2bf(f.x); o.y = f2bf(f.y); o.z = f2bf(f.z); o.w = f2bf(f.w);
    *(ushort4*)(dst + i) = o;
}

// ---------------- kernel 2: t[m][r] = sum_k x[m,k] * A[r,k]  (q and v) ------
__global__ void k_lora_t(const float* __restrict__ x,
                         const float* __restrict__ Aq,
                         const float* __restrict__ Av,
                         float* __restrict__ tq,
                         float* __restrict__ tv) {
    const int wave = threadIdx.x >> 5;
    const int lane = threadIdx.x & 31;
    const int m = blockIdx.x * 8 + wave;           // one row per wave
    const float* xr = x + (size_t)m * kD;
    float aq[kR] = {0.f, 0.f, 0.f, 0.f};
    float av[kR] = {0.f, 0.f, 0.f, 0.f};
    for (int k = lane; k < kD; k += 32) {
        const float xv = xr[k];
#pragma unroll
        for (int r = 0; r < kR; ++r) {
            aq[r] = fmaf(xv, Aq[r * kD + k], aq[r]);
            av[r] = fmaf(xv, Av[r * kD + k], av[r]);
        }
    }
#pragma unroll
    for (int r = 0; r < kR; ++r) {
#pragma unroll
        for (int off = 16; off > 0; off >>= 1) {
            aq[r] += __shfl_xor(aq[r], off, 32);
            av[r] += __shfl_xor(av[r], off, 32);
        }
    }
    if (lane == 0) {
#pragma unroll
        for (int r = 0; r < kR; ++r) {
            tq[(size_t)m * kR + r] = aq[r];
            tv[(size_t)m * kR + r] = av[r];
        }
    }
}

// ---------------- kernel 3: main bf16 WMMA GEMM + bias + LoRA epilogue ------
// Block: 256 threads (8 waves), 128(M) x 128(N) tile, K-step 32.
// Wave grid 4x2: wave computes 32(M) x 64(N) = 2x4 WMMA tiles.
__global__ __launch_bounds__(256)
void k_gemm_wmma(const unsigned short* __restrict__ xb,   // [65536][768] bf16
                 const unsigned short* __restrict__ wb,   // [2304][768]  bf16
                 const float* __restrict__ bias,          // [2304]
                 const float* __restrict__ Bq,            // [768][4]
                 const float* __restrict__ Bv,            // [768][4]
                 const float* __restrict__ tq,            // [65536][4]
                 const float* __restrict__ tv,            // [65536][4]
                 const float* __restrict__ s0p,
                 float* __restrict__ out) {               // [65536][2304]
    const int tid  = threadIdx.x;
    const int lane = tid & 31;
    const int wave = tid >> 5;
    const int wm   = wave & 3;        // M strip (x32)
    const int wn   = wave >> 2;       // N strip (x64)
    const int m0   = blockIdx.x * 128;
    const int n0   = blockIdx.y * 128;

    // staging: thread -> (row, 16-half chunk); 256 threads cover 128 rows x 32 halves
    const int sr = tid >> 1;
    const int sh = (tid & 1) * 16;
    const size_t gA0 = (size_t)(m0 + sr) * kD + sh;
    const size_t gB0 = (size_t)(n0 + sr) * kD + sh;

    v8f acc[2][4];
#pragma unroll
    for (int mt = 0; mt < 2; ++mt)
#pragma unroll
        for (int nt = 0; nt < 4; ++nt) {
            v8f z = {0.f, 0.f, 0.f, 0.f, 0.f, 0.f, 0.f, 0.f};
            acc[mt][nt] = z;
        }

    // fragment addresses (halves), constant across K-steps
    const int arow  = (wm * 32 + (lane & 15)) * LDA + ((lane >> 4) * 8);
    const int brow0 = (wn * 64 + (lane & 15)) * LDA + ((lane >> 4) * 8);
    constexpr int kSteps = kD / 32;   // 24

#if HAVE_ASYNC_LDS
    // ---- double-buffered LDS fed by GLOBAL_LOAD_ASYNC_TO_LDS_B128 ----------
    __shared__ __align__(16) unsigned short lA[2][128 * LDA];
    __shared__ __align__(16) unsigned short lB[2][128 * LDA];

    const int soff = sr * LDA + sh;
    // prologue: fill buffer 0 with K-step 0 (4 async b128 per thread)
    async_b128(&lA[0][soff],     xb + gA0);
    async_b128(&lA[0][soff + 8], xb + gA0 + 8);
    async_b128(&lB[0][soff],     wb + gB0);
    async_b128(&lB[0][soff + 8], wb + gB0 + 8);

    for (int i = 0; i < kSteps; ++i) {
        const int cur = i & 1;
        wait_async0();                 // this thread's copies into buf[cur] done
        __syncthreads();               // everyone's copies done; prev compute done

        if (i + 1 < kSteps) {          // overlap next tile's copy with WMMAs
            const int nb = cur ^ 1;
            const int ko = (i + 1) * 32;
            async_b128(&lA[nb][soff],     xb + gA0 + ko);
            async_b128(&lA[nb][soff + 8], xb + gA0 + ko + 8);
            async_b128(&lB[nb][soff],     wb + gB0 + ko);
            async_b128(&lB[nb][soff + 8], wb + gB0 + ko + 8);
        }

        const unsigned short* As = lA[cur];
        const unsigned short* Bs = lB[cur];

        FragAB a0, a1;
        a0.u[0] = *(const uint4*)(As + arow);
        a0.u[1] = *(const uint4*)(As + arow + 16);
        a1.u[0] = *(const uint4*)(As + arow + 16 * LDA);
        a1.u[1] = *(const uint4*)(As + arow + 16 * LDA + 16);

#pragma unroll
        for (int nt = 0; nt < 4; ++nt) {
            FragAB b;
            const int br = brow0 + nt * 16 * LDA;
            b.u[0] = *(const uint4*)(Bs + br);
            b.u[1] = *(const uint4*)(Bs + br + 16);
            acc[0][nt] = __builtin_amdgcn_wmma_f32_16x16x32_bf16(
                false, a0.v, false, b.v, (short)0, acc[0][nt], false, false);
            acc[1][nt] = __builtin_amdgcn_wmma_f32_16x16x32_bf16(
                false, a1.v, false, b.v, (short)0, acc[1][nt], false, false);
        }
    }
#else
    // ---- fallback: register-staged single-buffer LDS ---------
    __shared__ __align__(16) unsigned short lA1[128 * LDA];
    __shared__ __align__(16) unsigned short lB1[128 * LDA];

    uint4 ra0 = *(const uint4*)(xb + gA0);
    uint4 ra1 = *(const uint4*)(xb + gA0 + 8);
    uint4 rb0 = *(const uint4*)(wb + gB0);
    uint4 rb1 = *(const uint4*)(wb + gB0 + 8);

    for (int i = 0; i < kSteps; ++i) {
        const int k0 = i * 32;
        __syncthreads();
        *(uint4*)(&lA1[sr * LDA + sh])     = ra0;
        *(uint4*)(&lA1[sr * LDA + sh + 8]) = ra1;
        *(uint4*)(&lB1[sr * LDA + sh])     = rb0;
        *(uint4*)(&lB1[sr * LDA + sh + 8]) = rb1;
        __syncthreads();

        if (k0 + 32 < kD) {
            ra0 = *(const uint4*)(xb + gA0 + k0 + 32);
            ra1 = *(const uint4*)(xb + gA0 + k0 + 40);
            rb0 = *(const uint4*)(wb + gB0 + k0 + 32);
            rb1 = *(const uint4*)(wb + gB0 + k0 + 40);
        }

        FragAB a0, a1;
        a0.u[0] = *(const uint4*)(&lA1[arow]);
        a0.u[1] = *(const uint4*)(&lA1[arow + 16]);
        a1.u[0] = *(const uint4*)(&lA1[arow + 16 * LDA]);
        a1.u[1] = *(const uint4*)(&lA1[arow + 16 * LDA + 16]);

#pragma unroll
        for (int nt = 0; nt < 4; ++nt) {
            FragAB b;
            const int br = brow0 + nt * 16 * LDA;
            b.u[0] = *(const uint4*)(&lB1[br]);
            b.u[1] = *(const uint4*)(&lB1[br + 16]);
            acc[0][nt] = __builtin_amdgcn_wmma_f32_16x16x32_bf16(
                false, a0.v, false, b.v, (short)0, acc[0][nt], false, false);
            acc[1][nt] = __builtin_amdgcn_wmma_f32_16x16x32_bf16(
                false, a1.v, false, b.v, (short)0, acc[1][nt], false, false);
        }
    }
#endif

    // ---------------- epilogue: bias + LoRA(q/v) + store ----------------
    const float s0 = s0p[0];
    // column region is uniform per block: 768 and 1536 are multiples of 128
    const int region = (n0 >= 2 * kD) ? 2 : ((n0 < kD) ? 0 : 1);

#pragma unroll
    for (int mt = 0; mt < 2; ++mt) {
#pragma unroll
        for (int nt = 0; nt < 4; ++nt) {
            const int n     = n0 + wn * 64 + nt * 16 + (lane & 15);
            const int mbase = m0 + wm * 32 + mt * 16 + ((lane >> 4) << 3);
            const float bz  = bias[n];
            float4 bw = make_float4(0.f, 0.f, 0.f, 0.f);
            if (region == 0)      bw = *(const float4*)(Bq + (size_t)n * kR);
            else if (region == 2) bw = *(const float4*)(Bv + (size_t)(n - 2 * kD) * kR);
            const v8f c = acc[mt][nt];
#pragma unroll
            for (int i = 0; i < 8; ++i) {
                const int m = mbase + i;
                float v = c[i] + bz;
                if (region != 1) {
                    const float4 t = (region == 0)
                        ? *(const float4*)(tq + (size_t)m * kR)
                        : *(const float4*)(tv + (size_t)m * kR);
                    v += s0 * (t.x * bw.x + t.y * bw.y + t.z * bw.z + t.w * bw.w);
                }
                out[(size_t)m * kN + n] = v;
            }
        }
    }
}

// ---------------- host-side launch ----------------
extern "C" void kernel_launch(void* const* d_in, const int* in_sizes, int n_in,
                              void* d_out, int out_size, void* d_ws, size_t ws_size,
                              hipStream_t stream) {
    const float* x  = (const float*)d_in[0];
    const float* W  = (const float*)d_in[1];
    const float* b  = (const float*)d_in[2];
    const float* Aq = (const float*)d_in[3];
    const float* Bq = (const float*)d_in[4];
    const float* Av = (const float*)d_in[5];
    const float* Bv = (const float*)d_in[6];
    const float* s0 = (const float*)d_in[7];
    float* out = (float*)d_out;

    // workspace layout
    char* ws = (char*)d_ws;
    const size_t xb_bytes = (size_t)kM * kD * 2;   // 100,663,296
    const size_t wb_bytes = (size_t)kN * kD * 2;   //   3,538,944
    unsigned short* xb = (unsigned short*)ws;
    unsigned short* wb = (unsigned short*)(ws + xb_bytes);
    float* tq = (float*)(ws + xb_bytes + wb_bytes);
    float* tv = tq + (size_t)kM * kR;

    // 1) pack x and W to bf16 (exact multiples of 1024 elements)
    k_convert_bf16<<<(unsigned)((size_t)kM * kD / 4 / 256), 256, 0, stream>>>(x, xb);
    k_convert_bf16<<<(unsigned)((size_t)kN * kD / 4 / 256), 256, 0, stream>>>(W, wb);

    // 2) LoRA down-projections t = x @ A^T (q and v)
    k_lora_t<<<kM / 8, 256, 0, stream>>>(x, Aq, Av, tq, tv);

    // 3) main WMMA GEMM with fused bias + LoRA epilogue
    dim3 grid(kM / 128, kN / 128);   // 512 x 18
    k_gemm_wmma<<<grid, 256, 0, stream>>>(xb, wb, b, Bq, Bv, tq, tv, s0, out);
}
